// net_47304769798685
// MI455X (gfx1250) — matrix operand order repaired and checked by reading, not compiled, not run on hardware
//
#include <hip/hip_runtime.h>

#define N_NODES 100000
#define N_EDGES 6400000
#define F_IN    73

typedef __attribute__((ext_vector_type(16))) _Float16 v16h;
typedef __attribute__((ext_vector_type(8)))  float    v8f;

__device__ __forceinline__ float lrelu(float v, float s) { return v > 0.f ? v : s * v; }

__device__ __forceinline__ v8f wmma_f16(v16h a, v16h b, v8f c) {
  return __builtin_amdgcn_wmma_f32_16x16x32_f16(false, a, false, b, (short)0, c, false, false);
}

// 16-bit A-matrix 16x32 fragment (contiguous k, matching B swizzle): lanes 0-15 -> M=lane,
// K 0..15; lanes 16-31 -> M=lane-16, K 16..31.
__device__ __forceinline__ v16h load_a_frag(const _Float16* As, int ld, int lane, int kbase) {
  int m = lane & 15, hi = lane >> 4;
  union { v16h v; _Float16 h[16]; } r;
  const _Float16* row = As + m * ld + kbase;
#pragma unroll
  for (int j = 0; j < 8; ++j) {
    int kk = (hi << 4) + (j << 1);
    r.h[2 * j]     = row[kk];
    r.h[2 * j + 1] = row[kk + 1];
  }
  return r.v;
}

// B fragments are pre-swizzled in LDS: [frag][lane][8 dwords] contiguous.
__device__ __forceinline__ v16h load_b_frag(const unsigned* Bsw, int frag, int lane) {
  const unsigned* p = Bsw + (frag * 32 + lane) * 8;
  union { v16h v; unsigned u[8]; } r;
#pragma unroll
  for (int j = 0; j < 8; ++j) r.u[j] = p[j];
  return r.v;
}

// ---------------- init: h <- bias, emax <- -inf, den <- 0, stats <- 0 -------
__global__ void k_init(float* __restrict__ hbuf, float* __restrict__ emax,
                       float* __restrict__ den, float* __restrict__ stats,
                       const float* __restrict__ gat_bias) {
  int i = blockIdx.x * 256 + threadIdx.x;
  if (i < N_NODES * 128) hbuf[i] = gat_bias[i & 127];
  if (i < N_NODES * 4) { emax[i] = __uint_as_float(0xff800000u); den[i] = 0.f; }
  if (i < 2560) stats[i] = 0.f;
}

// ---------------- feat = x @ W_gat  (WMMA f16->f32, K padded 73->96) --------
__global__ __launch_bounds__(256)
void k_gemm_feat(const float* __restrict__ x, const float* __restrict__ Wg,
                 float* __restrict__ feat) {
  __shared__ _Float16 As[128 * 96];        // 24 KB
  __shared__ unsigned Bsw[3 * 8 * 32 * 8]; // 24 KB
  const int tid = threadIdx.x;
  const int n0 = blockIdx.x * 128;

  for (int idx = tid; idx < 128 * 96; idx += 256) {
    int r = idx / 96, c = idx - r * 96;
    int g = n0 + r;
    float v = (g < N_NODES && c < F_IN) ? x[g * F_IN + c] : 0.f;
    As[idx] = (_Float16)v;
  }
  for (int idx = tid; idx < 3 * 8 * 32 * 8; idx += 256) {
    int dw = idx & 7, lane = (idx >> 3) & 31, nt = (idx >> 8) & 7, ks = idx >> 11;
    int hi = lane >> 4, n = (nt << 4) + (lane & 15);
    int k0 = ks * 32 + hi * 16 + dw * 2;
    union { unsigned u; _Float16 h[2]; } p;
    p.h[0] = (_Float16)((k0 < F_IN) ? Wg[k0 * 128 + n] : 0.f);
    p.h[1] = (_Float16)((k0 + 1 < F_IN) ? Wg[(k0 + 1) * 128 + n] : 0.f);
    Bsw[idx] = p.u;
  }
  __syncthreads();

  const int wave = tid >> 5, lane = tid & 31;
  const int m0 = wave * 16;
  v8f zero = {0.f, 0.f, 0.f, 0.f, 0.f, 0.f, 0.f, 0.f};
  v8f acc[8];
#pragma unroll
  for (int nt = 0; nt < 8; ++nt) acc[nt] = zero;
#pragma unroll
  for (int ks = 0; ks < 3; ++ks) {
    v16h a = load_a_frag(As + m0 * 96, 96, lane, ks * 32);
#pragma unroll
    for (int nt = 0; nt < 8; ++nt) {
      v16h b = load_b_frag(Bsw, ks * 8 + nt, lane);
      acc[nt] = wmma_f16(a, b, acc[nt]);
    }
  }
  const int hi = lane >> 4, nlo = lane & 15;
#pragma unroll
  for (int nt = 0; nt < 8; ++nt) {
    union { v8f v; float f[8]; } u; u.v = acc[nt];
    int col = nt * 16 + nlo;
#pragma unroll
    for (int r = 0; r < 8; ++r) {
      int g = n0 + m0 + r + hi * 8;
      if (g < N_NODES) feat[g * 128 + col] = u.f[r];
    }
  }
}

// ---------------- el/er = sum_d feat*attn  (wave per node) ------------------
__global__ __launch_bounds__(256)
void k_attn_dots(const float* __restrict__ feat, const float* __restrict__ attn_l,
                 const float* __restrict__ attn_r, float* __restrict__ el,
                 float* __restrict__ er) {
  int wave = threadIdx.x >> 5, lane = threadIdx.x & 31;
  int n = blockIdx.x * 8 + wave;
  if (n >= N_NODES) return;
  const float4 f = *(const float4*)(feat + n * 128 + lane * 4);
  int head = lane >> 3;
  int doff = (lane & 7) * 4;
  const float4 al = *(const float4*)(attn_l + head * 32 + doff);
  const float4 ar = *(const float4*)(attn_r + head * 32 + doff);
  float pl = f.x * al.x + f.y * al.y + f.z * al.z + f.w * al.w;
  float pr = f.x * ar.x + f.y * ar.y + f.z * ar.z + f.w * ar.w;
#pragma unroll
  for (int m = 1; m < 8; m <<= 1) {
    pl += __shfl_xor(pl, m, 32);
    pr += __shfl_xor(pr, m, 32);
  }
  if ((lane & 7) == 0) { el[n * 4 + head] = pl; er[n * 4 + head] = pr; }
}

// ---------------- edge pass 1: segment max ---------------------------------
__device__ __forceinline__ void atomicMaxF(float* addr, float v) {
  if (v >= 0.f) atomicMax((int*)addr, __float_as_int(v));
  else          atomicMin((unsigned int*)addr, __float_as_uint(v));
}

__global__ void k_edge_max(const int* __restrict__ src, const int* __restrict__ dst,
                           const float* __restrict__ el, const float* __restrict__ er,
                           float* __restrict__ emax) {
  int e = blockIdx.x * 256 + threadIdx.x;
  if (e >= N_EDGES) return;
  int s = src[e], d = dst[e];
  float4 a = *(const float4*)(el + s * 4);
  float4 b = *(const float4*)(er + d * 4);
  atomicMaxF(&emax[d * 4 + 0], lrelu(a.x + b.x, 0.2f));
  atomicMaxF(&emax[d * 4 + 1], lrelu(a.y + b.y, 0.2f));
  atomicMaxF(&emax[d * 4 + 2], lrelu(a.z + b.z, 0.2f));
  atomicMaxF(&emax[d * 4 + 3], lrelu(a.w + b.w, 0.2f));
}

// ---------------- edge pass 2: exp + segment sum ---------------------------
__global__ void k_edge_expsum(const int* __restrict__ src, const int* __restrict__ dst,
                              const float* __restrict__ el, const float* __restrict__ er,
                              const float* __restrict__ emax, float* __restrict__ den) {
  int e = blockIdx.x * 256 + threadIdx.x;
  if (e >= N_EDGES) return;
  int s = src[e], d = dst[e];
  float4 a = *(const float4*)(el + s * 4);
  float4 b = *(const float4*)(er + d * 4);
  float4 m = *(const float4*)(emax + d * 4);
  atomicAdd(&den[d * 4 + 0], __expf(lrelu(a.x + b.x, 0.2f) - m.x));
  atomicAdd(&den[d * 4 + 1], __expf(lrelu(a.y + b.y, 0.2f) - m.y));
  atomicAdd(&den[d * 4 + 2], __expf(lrelu(a.z + b.z, 0.2f) - m.z));
  atomicAdd(&den[d * 4 + 3], __expf(lrelu(a.w + b.w, 0.2f) - m.w));
}

// ---------------- edge pass 3: h[dst] += alpha * feat[src] (wave per edge) --
__global__ void k_edge_agg(const int* __restrict__ src, const int* __restrict__ dst,
                           const float* __restrict__ el, const float* __restrict__ er,
                           const float* __restrict__ emax, const float* __restrict__ den,
                           const float* __restrict__ feat, float* __restrict__ hbuf) {
  int gid = blockIdx.x * 256 + threadIdx.x;
  int e = gid >> 5;
  int lane = gid & 31;
  if (e >= N_EDGES) return;
  int s = src[e], d = dst[e];
  if (e + 8 < N_EDGES) __builtin_prefetch(feat + src[e + 8] * 128 + lane * 4, 0, 1);
  int head = lane >> 3;
  float ev = lrelu(el[s * 4 + head] + er[d * 4 + head], 0.2f);
  float ex = __expf(ev - emax[d * 4 + head]);
  float alpha = ex / fmaxf(den[d * 4 + head], 1e-9f);
  const float4 f = *(const float4*)(feat + s * 128 + lane * 4);
  float* hp = hbuf + d * 128 + lane * 4;
  atomicAdd(hp + 0, f.x * alpha);
  atomicAdd(hp + 1, f.y * alpha);
  atomicAdd(hp + 2, f.z * alpha);
  atomicAdd(hp + 3, f.w * alpha);
}

// ---------------- BN stats: per-channel sum / sumsq ------------------------
__global__ void k_stats(const float* __restrict__ in, float* __restrict__ sum,
                        float* __restrict__ sumsq, int C, int total) {
  int tid = threadIdx.x;
  int stride = gridDim.x * 256;   // multiple of C for C in {2,4,8,32,128}
  float s = 0.f, q = 0.f;
  for (int i = blockIdx.x * 256 + tid; i < total; i += stride) {
    float v = in[i]; s += v; q += v * v;
  }
  __shared__ float ls[256], lq[256];
  ls[tid] = s; lq[tid] = q;
  __syncthreads();
  if (tid < C) {
    for (int j = tid + C; j < 256; j += C) { s += ls[j]; q += lq[j]; }
    atomicAdd(&sum[tid], s);
    atomicAdd(&sumsq[tid], q);
  }
}

__global__ void k_finalize(const float* __restrict__ sum, const float* __restrict__ sumsq,
                           const float* __restrict__ g, const float* __restrict__ b,
                           float* __restrict__ scale, float* __restrict__ shift,
                           int C, float invN) {
  int c = threadIdx.x;
  if (c >= C) return;
  float mean = sum[c] * invN;
  float var  = sumsq[c] * invN - mean * mean;
  float inv  = rsqrtf(var + 1e-5f);
  float sc   = g[c] * inv;
  scale[c] = sc;
  shift[c] = b[c] - mean * sc;
}

// ---------------- z2 = lrelu(bn1(h)) @ fc1_w + b  (WMMA, K=128, Nc=32) -----
__global__ __launch_bounds__(256)
void k_mlp1(const float* __restrict__ hbuf, const float* __restrict__ scale,
            const float* __restrict__ shift, const float* __restrict__ W,
            const float* __restrict__ bias, float* __restrict__ z2) {
  __shared__ _Float16 As[128 * 128];       // 32 KB
  __shared__ unsigned Bsw[4 * 2 * 32 * 8]; // 8 KB
  const int tid = threadIdx.x;
  const int n0 = blockIdx.x * 128;

  for (int idx = tid; idx < 128 * 128; idx += 256) {
    int r = idx >> 7, c = idx & 127;
    int g = n0 + r;
    float v = 0.f;
    if (g < N_NODES) v = lrelu(scale[c] * hbuf[g * 128 + c] + shift[c], 0.1f);
    As[idx] = (_Float16)v;
  }
  for (int idx = tid; idx < 4 * 2 * 32 * 8; idx += 256) {
    int dw = idx & 7, lane = (idx >> 3) & 31, nt = (idx >> 8) & 1, ks = idx >> 9;
    int hi = lane >> 4, n = (nt << 4) + (lane & 15);
    int k0 = ks * 32 + hi * 16 + dw * 2;
    union { unsigned u; _Float16 h[2]; } p;
    p.h[0] = (_Float16)W[k0 * 32 + n];
    p.h[1] = (_Float16)W[(k0 + 1) * 32 + n];
    Bsw[idx] = p.u;
  }
  __syncthreads();

  const int wave = tid >> 5, lane = tid & 31;
  const int m0 = wave * 16;
  v8f zero = {0.f, 0.f, 0.f, 0.f, 0.f, 0.f, 0.f, 0.f};
  v8f acc0 = zero, acc1 = zero;
#pragma unroll
  for (int ks = 0; ks < 4; ++ks) {
    v16h a  = load_a_frag(As + m0 * 128, 128, lane, ks * 32);
    v16h b0 = load_b_frag(Bsw, ks * 2 + 0, lane);
    v16h b1 = load_b_frag(Bsw, ks * 2 + 1, lane);
    acc0 = wmma_f16(a, b0, acc0);
    acc1 = wmma_f16(a, b1, acc1);
  }
  const int hi = lane >> 4, nlo = lane & 15;
  float b0v = bias[nlo], b1v = bias[16 + nlo];
  union { v8f v; float f[8]; } u0, u1;
  u0.v = acc0; u1.v = acc1;
#pragma unroll
  for (int r = 0; r < 8; ++r) {
    int g = n0 + m0 + r + hi * 8;
    if (g < N_NODES) {
      z2[g * 32 + nlo]      = u0.f[r] + b0v;
      z2[g * 32 + 16 + nlo] = u1.f[r] + b1v;
    }
  }
}

// ---------------- small FC layers: out = lrelu(bn(in)) @ W + b -------------
template <int CIN, int COUT>
__global__ void k_fc(const float* __restrict__ in, const float* __restrict__ scale,
                     const float* __restrict__ shift, const float* __restrict__ W,
                     const float* __restrict__ bias, float* __restrict__ out, int n) {
  __shared__ float ws[CIN * COUT];
  __shared__ float bs[COUT];
  for (int i = threadIdx.x; i < CIN * COUT; i += 256) ws[i] = W[i];
  for (int i = threadIdx.x; i < COUT; i += 256) bs[i] = bias[i];
  __syncthreads();
  int idx = blockIdx.x * 256 + threadIdx.x;
  if (idx >= n) return;
  float a[CIN];
#pragma unroll
  for (int c = 0; c < CIN; ++c)
    a[c] = lrelu(scale[c] * in[idx * CIN + c] + shift[c], 0.1f);
#pragma unroll
  for (int j = 0; j < COUT; ++j) {
    float s = bs[j];
#pragma unroll
    for (int c = 0; c < CIN; ++c) s += a[c] * ws[c * COUT + j];
    out[idx * COUT + j] = s;
  }
}

// ---------------- bn5 + lrelu + softmax(2) ---------------------------------
__global__ void k_softmax2(const float* __restrict__ z5, const float* __restrict__ scale,
                           const float* __restrict__ shift, float* __restrict__ out, int n) {
  int i = blockIdx.x * 256 + threadIdx.x;
  if (i >= n) return;
  float a0 = lrelu(scale[0] * z5[i * 2 + 0] + shift[0], 0.1f);
  float a1 = lrelu(scale[1] * z5[i * 2 + 1] + shift[1], 0.1f);
  float m = fmaxf(a0, a1);
  float e0 = __expf(a0 - m), e1 = __expf(a1 - m);
  float inv = 1.f / (e0 + e1);
  out[i * 2 + 0] = e0 * inv;
  out[i * 2 + 1] = e1 * inv;
}

extern "C" void kernel_launch(void* const* d_in, const int* in_sizes, int n_in,
                              void* d_out, int out_size, void* d_ws, size_t ws_size,
                              hipStream_t stream) {
  (void)in_sizes; (void)n_in; (void)out_size; (void)ws_size;
  const float* x        = (const float*)d_in[0];
  const int*   src      = (const int*)d_in[1];
  const int*   dst      = (const int*)d_in[2];
  const float* Wg       = (const float*)d_in[3];
  const float* attn_l   = (const float*)d_in[4];
  const float* attn_r   = (const float*)d_in[5];
  const float* gat_bias = (const float*)d_in[6];
  const float* fc1_w = (const float*)d_in[7];  const float* fc1_b = (const float*)d_in[8];
  const float* fc2_w = (const float*)d_in[9];  const float* fc2_b = (const float*)d_in[10];
  const float* fc3_w = (const float*)d_in[11]; const float* fc3_b = (const float*)d_in[12];
  const float* fc4_w = (const float*)d_in[13]; const float* fc4_b = (const float*)d_in[14];
  const float* bng[5] = {(const float*)d_in[15], (const float*)d_in[17], (const float*)d_in[19],
                         (const float*)d_in[21], (const float*)d_in[23]};
  const float* bnb[5] = {(const float*)d_in[16], (const float*)d_in[18], (const float*)d_in[20],
                         (const float*)d_in[22], (const float*)d_in[24]};

  float* w = (float*)d_ws;
  float* feat = w;  w += N_NODES * 128;
  float* hbuf = w;  w += N_NODES * 128;
  float* el   = w;  w += N_NODES * 4;
  float* er   = w;  w += N_NODES * 4;
  float* emax = w;  w += N_NODES * 4;
  float* den  = w;  w += N_NODES * 4;
  float* z2   = w;  w += N_NODES * 32;
  float* z3   = w;  w += N_NODES * 8;
  float* z4   = w;  w += N_NODES * 4;
  float* z5   = w;  w += N_NODES * 2;
  float* stats = w; // 5 layers * 512 floats: [sum(128)|sumsq(128)|scale(128)|shift(128)]

  float* sum[5], *sq[5], *sc[5], *sh[5];
  for (int i = 0; i < 5; ++i) {
    sum[i] = stats + i * 512;
    sq[i]  = sum[i] + 128;
    sc[i]  = sum[i] + 256;
    sh[i]  = sum[i] + 384;
  }
  const int Cs[5] = {128, 32, 8, 4, 2};
  const float invN = 1.f / (float)N_NODES;

  // GAT stage
  k_init<<<(N_NODES * 128 + 255) / 256, 256, 0, stream>>>(hbuf, emax, den, stats, gat_bias);
  k_gemm_feat<<<(N_NODES + 127) / 128, 256, 0, stream>>>(x, Wg, feat);
  k_attn_dots<<<(N_NODES + 7) / 8, 256, 0, stream>>>(feat, attn_l, attn_r, el, er);
  k_edge_max<<<(N_EDGES + 255) / 256, 256, 0, stream>>>(src, dst, el, er, emax);
  k_edge_expsum<<<(N_EDGES + 255) / 256, 256, 0, stream>>>(src, dst, el, er, emax, den);
  k_edge_agg<<<N_EDGES / 8, 256, 0, stream>>>(src, dst, el, er, emax, den, feat, hbuf);

  // BN1 + fc1 (WMMA)
  k_stats<<<2048, 256, 0, stream>>>(hbuf, sum[0], sq[0], Cs[0], N_NODES * 128);
  k_finalize<<<1, 128, 0, stream>>>(sum[0], sq[0], bng[0], bnb[0], sc[0], sh[0], Cs[0], invN);
  k_mlp1<<<(N_NODES + 127) / 128, 256, 0, stream>>>(hbuf, sc[0], sh[0], fc1_w, fc1_b, z2);

  // BN2 + fc2
  k_stats<<<2048, 256, 0, stream>>>(z2, sum[1], sq[1], Cs[1], N_NODES * 32);
  k_finalize<<<1, 128, 0, stream>>>(sum[1], sq[1], bng[1], bnb[1], sc[1], sh[1], Cs[1], invN);
  k_fc<32, 8><<<(N_NODES + 255) / 256, 256, 0, stream>>>(z2, sc[1], sh[1], fc2_w, fc2_b, z3, N_NODES);

  // BN3 + fc3
  k_stats<<<1024, 256, 0, stream>>>(z3, sum[2], sq[2], Cs[2], N_NODES * 8);
  k_finalize<<<1, 128, 0, stream>>>(sum[2], sq[2], bng[2], bnb[2], sc[2], sh[2], Cs[2], invN);
  k_fc<8, 4><<<(N_NODES + 255) / 256, 256, 0, stream>>>(z3, sc[2], sh[2], fc3_w, fc3_b, z4, N_NODES);

  // BN4 + fc4
  k_stats<<<512, 256, 0, stream>>>(z4, sum[3], sq[3], Cs[3], N_NODES * 4);
  k_finalize<<<1, 128, 0, stream>>>(sum[3], sq[3], bng[3], bnb[3], sc[3], sh[3], Cs[3], invN);
  k_fc<4, 2><<<(N_NODES + 255) / 256, 256, 0, stream>>>(z4, sc[3], sh[3], fc4_w, fc4_b, z5, N_NODES);

  // BN5 + softmax
  k_stats<<<512, 256, 0, stream>>>(z5, sum[4], sq[4], Cs[4], N_NODES * 2);
  k_finalize<<<1, 128, 0, stream>>>(sum[4], sq[4], bng[4], bnb[4], sc[4], sh[4], Cs[4], invN);
  k_softmax2<<<(N_NODES + 255) / 256, 256, 0, stream>>>(z5, sc[4], sh[4], (float*)d_out, N_NODES);
}
